// NeuroplasticityLayer_62474594287802
// MI455X (gfx1250) — compile-verified, matrix operand order checked by reading, not run on hardware
//
#include <hip/hip_runtime.h>
#include <math.h>

#define H 2048
#define NROWS 16384            // B*S = 8*2048
#define KSPLIT 4
#define KCHUNK 64              // two 16x16x32 K-steps per LDS stage
#define LDSTR 72               // 64 bf16 payload + 8 pad -> 144B rows = 36 banks, conflict-free b128 frags

typedef __attribute__((ext_vector_type(16))) __bf16 v16bf;
typedef __attribute__((ext_vector_type(8)))  __bf16 v8bf;
typedef __attribute__((ext_vector_type(8)))  float  v8f;
typedef __attribute__((ext_vector_type(4)))  float  v4f;

// packed f32->bf16 (lowers to v_cvt_pk_bf16_f32)
__device__ __forceinline__ v8bf cvt8(v4f a, v4f b) {
    v8bf r;
    #pragma unroll
    for (int j = 0; j < 4; ++j) { r[j] = (__bf16)a[j]; r[j + 4] = (__bf16)b[j]; }
    return r;
}

// CDNA5 async copy: global -> LDS, 16B per lane, tracked by ASYNCcnt
__device__ __forceinline__ void async_ld_b128(unsigned ldsoff, unsigned long long gaddr) {
    asm volatile("global_load_async_to_lds_b128 %0, %1, off"
                 :: "v"(ldsoff), "v"(gaddr) : "memory");
}
__device__ __forceinline__ void wait_async0() {
    asm volatile("s_wait_asynccnt 0" ::: "memory");
}

// 16x32 bf16 operand fragment per ISA layout:
// lanes 0-15: K=0..7,16..23 ; lanes 16-31: K=8..15,24..31 ; row = lane&15.
__device__ __forceinline__ v16bf load_frag(const __bf16* rowptr, int lane) {
    const int kOff = (lane < 16) ? 0 : 8;
    v8bf lo = *(const v8bf*)(rowptr + kOff);
    v8bf hi = *(const v8bf*)(rowptr + kOff + 16);
    return __builtin_shufflevector(lo, hi, 0,1,2,3,4,5,6,7,8,9,10,11,12,13,14,15);
}

// ---------------- W_sum = bf16(fixed_w + plastic) ----------------
__global__ __launch_bounds__(256)
void wsum_bf_kernel(const float* __restrict__ fw, const float* __restrict__ pw,
                    __bf16* __restrict__ out) {
    const size_t i = (size_t)blockIdx.x * blockDim.x + threadIdx.x;  // over H*H/8
    const v4f* a = (const v4f*)fw;
    const v4f* b = (const v4f*)pw;
    v4f s0 = a[2 * i] + b[2 * i];
    v4f s1 = a[2 * i + 1] + b[2 * i + 1];
    ((v8bf*)out)[i] = cvt8(s0, s1);
}

// ---------------- xT_bf[h][n] = bf16(x[n][h])  (tiled transpose) ----------------
__global__ __launch_bounds__(256)
void xt_bf_kernel(const float* __restrict__ x, __bf16* __restrict__ xt) {
    __shared__ float tile[64][65];
    const int t = threadIdx.x;
    const int nBase = blockIdx.x * 64;
    const int hBase = blockIdx.y * 64;
    const int cl = (t & 15) * 4, rw = t >> 4;
    #pragma unroll
    for (int p = 0; p < 4; ++p) {
        const int r = rw + p * 16;
        v4f v = *(const v4f*)(x + (size_t)(nBase + r) * H + hBase + cl);
        tile[r][cl] = v[0]; tile[r][cl + 1] = v[1];
        tile[r][cl + 2] = v[2]; tile[r][cl + 3] = v[3];
    }
    __syncthreads();
    const int cseg = (t & 7) * 8, hrow = t >> 3;
    #pragma unroll
    for (int p = 0; p < 2; ++p) {
        const int hr = hrow + p * 32;
        v8bf o;
        #pragma unroll
        for (int j = 0; j < 8; ++j) o[j] = (__bf16)tile[cseg + j][hr];
        *(v8bf*)(xt + (size_t)(hBase + hr) * NROWS + nBase + cseg) = o;
    }
}

// ---------------- GEMM1: combined[m,n] = sum_k x[m,k]*Wsum[n,k] + bias[n] ----------------
// A: f32 x, cvt on stage.  B: bf16 wsum, async global->LDS.
__global__ __launch_bounds__(256)
void gemm1_kernel(const float* __restrict__ x, const __bf16* __restrict__ wsum_bf,
                  const float* __restrict__ bias, float* __restrict__ out) {
    __shared__ __align__(16) __bf16 lA[2][128 * LDSTR];
    __shared__ __align__(16) __bf16 lB[2][128 * LDSTR];

    const int t = threadIdx.x;
    const int lane = t & 31, lane16 = lane & 15;
    const int wave = t >> 5;
    const int wm = wave >> 1;            // 0..3 : 32-row slice of M
    const int wn = wave & 1;             // 0..1 : 64-col slice of N
    const int mBase = blockIdx.y * 128;
    const int nBase = blockIdx.x * 128;

    v8f acc[2][4] = {};

    const int c0   = (t & 7) * 8;        // k-segment within 64-wide chunk
    const int row0 = t >> 3;             // 0..31, 4 passes of +32

    v4f ra[8];

    auto prefetchA = [&](int kB) {
        #pragma unroll
        for (int p = 0; p < 4; ++p) {
            const int r = row0 + p * 32;
            const float* gx = x + (size_t)(mBase + r) * H + kB + c0;
            ra[2*p] = *(const v4f*)gx; ra[2*p+1] = *(const v4f*)(gx + 4);
        }
    };
    auto stashA = [&](int buf) {
        #pragma unroll
        for (int p = 0; p < 4; ++p) {
            const int r = row0 + p * 32;
            *(v8bf*)(lA[buf] + r * LDSTR + c0) = cvt8(ra[2*p], ra[2*p+1]);
        }
    };
    auto asyncB = [&](int kB, int buf) {
        #pragma unroll
        for (int p = 0; p < 4; ++p) {
            const int r = row0 + p * 32;
            unsigned loff = (unsigned)(size_t)(lB[buf] + r * LDSTR + c0);
            unsigned long long g =
                (unsigned long long)(wsum_bf + (size_t)(nBase + r) * H + kB + c0);
            async_ld_b128(loff, g);
        }
    };
    auto compute = [&](int buf) {
        #pragma unroll
        for (int s = 0; s < 2; ++s) {
            v16bf af[2], bfr[4];
            #pragma unroll
            for (int ti = 0; ti < 2; ++ti)
                af[ti] = load_frag(lA[buf] + (wm*32 + ti*16 + lane16) * LDSTR + s*32, lane);
            #pragma unroll
            for (int tj = 0; tj < 4; ++tj)
                bfr[tj] = load_frag(lB[buf] + (wn*64 + tj*16 + lane16) * LDSTR + s*32, lane);
            #pragma unroll
            for (int ti = 0; ti < 2; ++ti)
                #pragma unroll
                for (int tj = 0; tj < 4; ++tj)
                    acc[ti][tj] = __builtin_amdgcn_wmma_f32_16x16x32_bf16(
                        false, af[ti], false, bfr[tj], (short)0, acc[ti][tj], false, false);
        }
    };

    prefetchA(0);
    asyncB(0, 0);
    stashA(0);
    wait_async0();
    __syncthreads();
    int buf = 0;
    const int NIT = H / KCHUNK;
    for (int i = 0; i < NIT; ++i) {
        if (i + 1 < NIT) { prefetchA((i + 1) * KCHUNK); asyncB((i + 1) * KCHUNK, buf ^ 1); }
        compute(buf);
        if (i + 1 < NIT) stashA(buf ^ 1);
        wait_async0();
        __syncthreads();
        buf ^= 1;
    }

    // Epilogue: VGPR r: lanes 0-15 hold M=r, lanes 16-31 hold M=r+8, N=lane&15
    #pragma unroll
    for (int ti = 0; ti < 2; ++ti) {
        const int mrow = mBase + wm * 32 + ti * 16 + ((lane < 16) ? 0 : 8);
        #pragma unroll
        for (int tj = 0; tj < 4; ++tj) {
            const int n = nBase + wn * 64 + tj * 16 + lane16;
            const float bv = bias[n];
            #pragma unroll
            for (int r = 0; r < 8; ++r)
                out[(size_t)(mrow + r) * H + n] = acc[ti][tj][r] + bv;
        }
    }
}

// ---------------- GEMM2: hebb[m,n] += sum_k x[k,m]*combined[k,n]  (split-K, atomics) ----------------
// A: bf16 xT (pre-transposed), async global->LDS.  B: f32 combined, transpose on LDS store.
__global__ __launch_bounds__(256)
void gemm2_kernel(const __bf16* __restrict__ xt, const float* __restrict__ comb,
                  float* __restrict__ hebb) {
    __shared__ __align__(16) __bf16 lA[2][128 * LDSTR];
    __shared__ __align__(16) __bf16 lB[2][128 * LDSTR];

    const int t = threadIdx.x;
    const int lane = t & 31, lane16 = lane & 15;
    const int wave = t >> 5;
    const int wm = wave >> 1;
    const int wn = wave & 1;
    const int mBase  = blockIdx.y * 128;                 // hebb row (h)
    const int nBase  = blockIdx.x * 128;                 // hebb col
    const int kStart = blockIdx.z * (NROWS / KSPLIT);

    v8f acc[2][4] = {};

    const int c0a = (t & 7) * 8;         // reduction-dim segment for async A
    const int r0a = t >> 3;              // 0..31, 4 passes of +32 (h rows)
    const int c0b = (t & 15) * 8;        // output-dim column segment 0..120
    const int n0  = t >> 4;              // 0..15 reduction rows, 4 passes of +16

    v4f rb[8];

    auto asyncA = [&](int kB, int buf) {
        #pragma unroll
        for (int p = 0; p < 4; ++p) {
            const int r = r0a + p * 32;
            unsigned loff = (unsigned)(size_t)(lA[buf] + r * LDSTR + c0a);
            unsigned long long g =
                (unsigned long long)(xt + (size_t)(mBase + r) * NROWS + kB + c0a);
            async_ld_b128(loff, g);
        }
    };
    auto prefetchB = [&](int kB) {
        #pragma unroll
        for (int p = 0; p < 4; ++p) {
            const int n = n0 + p * 16;
            const float* gc = comb + (size_t)(kB + n) * H + nBase + c0b;  // coalesced rows
            rb[2*p] = *(const v4f*)gc; rb[2*p+1] = *(const v4f*)(gc + 4);
        }
    };
    auto stashB = [&](int buf) {         // transpose on LDS store: [outcol][n]
        #pragma unroll
        for (int p = 0; p < 4; ++p) {
            const int n = n0 + p * 16;
            #pragma unroll
            for (int j = 0; j < 4; ++j) {
                lB[buf][(c0b + j    ) * LDSTR + n] = (__bf16)rb[2*p][j];
                lB[buf][(c0b + j + 4) * LDSTR + n] = (__bf16)rb[2*p+1][j];
            }
        }
    };
    auto compute = [&](int buf) {
        #pragma unroll
        for (int s = 0; s < 2; ++s) {
            v16bf af[2], bfr[4];
            #pragma unroll
            for (int ti = 0; ti < 2; ++ti)
                af[ti] = load_frag(lA[buf] + (wm*32 + ti*16 + lane16) * LDSTR + s*32, lane);
            #pragma unroll
            for (int tj = 0; tj < 4; ++tj)
                bfr[tj] = load_frag(lB[buf] + (wn*64 + tj*16 + lane16) * LDSTR + s*32, lane);
            #pragma unroll
            for (int ti = 0; ti < 2; ++ti)
                #pragma unroll
                for (int tj = 0; tj < 4; ++tj)
                    acc[ti][tj] = __builtin_amdgcn_wmma_f32_16x16x32_bf16(
                        false, af[ti], false, bfr[tj], (short)0, acc[ti][tj], false, false);
        }
    };

    prefetchB(kStart);
    asyncA(kStart, 0);
    stashB(0);
    wait_async0();
    __syncthreads();
    int buf = 0;
    const int NIT = (NROWS / KSPLIT) / KCHUNK;
    for (int i = 0; i < NIT; ++i) {
        const int kNext = kStart + (i + 1) * KCHUNK;
        if (i + 1 < NIT) { prefetchB(kNext); asyncA(kNext, buf ^ 1); }
        compute(buf);
        if (i + 1 < NIT) stashB(buf ^ 1);
        wait_async0();
        __syncthreads();
        buf ^= 1;
    }

    #pragma unroll
    for (int ti = 0; ti < 2; ++ti) {
        const int mrow = mBase + wm * 32 + ti * 16 + ((lane < 16) ? 0 : 8);
        #pragma unroll
        for (int tj = 0; tj < 4; ++tj) {
            const int n = nBase + wn * 64 + tj * 16 + lane16;
            #pragma unroll
            for (int r = 0; r < 8; ++r)
                atomicAdd(&hebb[(size_t)(mrow + r) * H + n], acc[ti][tj][r]);
        }
    }
}

// ---------------- new_w + norm^2 reduction ----------------
__global__ __launch_bounds__(256)
void finalize1_kernel(const float* __restrict__ plastic, const float* __restrict__ hebb,
                      const float* __restrict__ rate, const float* __restrict__ strength,
                      float* __restrict__ outw, float* __restrict__ normAcc) {
    const float factor = rate[0] * strength[0] * (1.0f / (float)NROWS);
    const size_t i = (size_t)blockIdx.x * blockDim.x + threadIdx.x;   // grid == H*H exactly
    const float nw = plastic[i] + factor * hebb[i];
    outw[i] = nw;
    float ss = nw * nw;
    #pragma unroll
    for (int off = 16; off > 0; off >>= 1) ss += __shfl_down(ss, off, 32);
    __shared__ float red[8];
    if ((threadIdx.x & 31) == 0) red[threadIdx.x >> 5] = ss;
    __syncthreads();
    if (threadIdx.x < 8) {
        float s = red[threadIdx.x];
        #pragma unroll
        for (int off = 4; off > 0; off >>= 1) s += __shfl_down(s, off, 8);
        if (threadIdx.x == 0) atomicAdd(normAcc, s);
    }
}

__global__ __launch_bounds__(256)
void finalize2_kernel(float* __restrict__ outw, const float* __restrict__ normAcc) {
    const float norm = sqrtf(*normAcc);
    const float scale = (norm > 1.0f) ? (1.0f / norm) : 1.0f;
    const size_t i = (size_t)blockIdx.x * blockDim.x + threadIdx.x;
    outw[i] *= scale;
}

extern "C" void kernel_launch(void* const* d_in, const int* in_sizes, int n_in,
                              void* d_out, int out_size, void* d_ws, size_t ws_size,
                              hipStream_t stream) {
    const float* x        = (const float*)d_in[0];   // [B,S,H] f32
    const float* plastic  = (const float*)d_in[1];   // [H,H]
    const float* rate     = (const float*)d_in[2];   // [1]
    const float* fixed_w  = (const float*)d_in[3];   // [H,H]
    const float* fixed_b  = (const float*)d_in[4];   // [H]
    const float* strength = (const float*)d_in[5];   // scalar

    float* out_combined = (float*)d_out;                               // NROWS*H
    float* out_w        = out_combined + (size_t)NROWS * H;            // H*H

    // workspace: xT_bf (64MB) | wsum_bf (8MB) | hebb f32 (16MB) | normAcc
    __bf16* xt_bf   = (__bf16*)d_ws;
    __bf16* wsum_bf = xt_bf + (size_t)H * NROWS;
    float*  hebb    = (float*)(wsum_bf + (size_t)H * H);
    float*  normAcc = hebb + (size_t)H * H;

    hipMemsetAsync(hebb, 0, (size_t)H * H * sizeof(float), stream);
    hipMemsetAsync(normAcc, 0, sizeof(float), stream);

    wsum_bf_kernel<<<(H * H / 8) / 256, 256, 0, stream>>>(fixed_w, plastic, wsum_bf);
    xt_bf_kernel<<<dim3(NROWS / 64, H / 64), 256, 0, stream>>>(x, xt_bf);

    gemm1_kernel<<<dim3(H / 128, NROWS / 128), 256, 0, stream>>>(x, wsum_bf, fixed_b,
                                                                 out_combined);

    gemm2_kernel<<<dim3(H / 128, H / 128, KSPLIT), 256, 0, stream>>>(xt_bf, out_combined, hebb);

    finalize1_kernel<<<(H * H) / 256, 256, 0, stream>>>(plastic, hebb, rate, strength,
                                                        out_w, normAcc);
    finalize2_kernel<<<(H * H) / 256, 256, 0, stream>>>(out_w, normAcc);
}
